// GCNN_uw_46755013984836
// MI455X (gfx1250) — matrix operand order repaired and checked by reading, not compiled, not run on hardware
//
#include <hip/hip_runtime.h>
#include <hip/hip_bf16.h>

// ---------------------------------------------------------------------------
// GCN 2-layer (GCNConv -> ReLU -> BN) x2 -> Linear, for gfx1250 (MI455X).
// GEMMs use CDNA5 WMMA (f32 16x16x4 path if the builtin exists, else the
// codegen-confirmed f16 16x16x32 path with f32 accumulation).
// Edge aggregation uses fire-and-forget global_atomic_add_f32; with h (<26MB)
// L2-resident on the 192MB L2, the scatter/gather stays on-chip.
// ---------------------------------------------------------------------------

typedef float    v2f  __attribute__((ext_vector_type(2)));
typedef float    v8f  __attribute__((ext_vector_type(8)));
typedef _Float16 v16h __attribute__((ext_vector_type(16)));

#define BN_EPS 1e-5f

#if __has_builtin(__builtin_amdgcn_wmma_f32_16x16x4_f32)
#define USE_F32_WMMA 1
#else
#define USE_F32_WMMA 0
#endif

// One 16x16 output tile per wave. blockDim.x = 32 * (N/16); wave w owns tileN=w.
// gridDim.x = ceil(M/16); block b owns tileM=b. A: MxK row-major, B: KxN row-major.
__global__ void gcn_gemm_wmma(const float* __restrict__ A, const float* __restrict__ B,
                              float* __restrict__ C, int M, int K, int N) {
  const int wave  = threadIdx.x >> 5;
  const int lane  = threadIdx.x & 31;
  const int tileM = blockIdx.x;
  const int tileN = wave;
  const int lm    = lane & 15;
  int mrow = tileM * 16 + lm;
  if (mrow >= M) mrow = M - 1;                 // clamp reads; stores guarded below
  const int ncol = tileN * 16 + lm;

  v8f c = {};

#if USE_F32_WMMA
  // 32-bit A 16x4 layout: lanes 0-15 hold K = {k, k+1}; lanes 16-31 hold {k+2, k+3}.
  const int hiK = (lane >> 4) << 1;            // 0 or 2
  const float* arow = A + (long)mrow * K;
  #pragma unroll 4
  for (int k = 0; k < K; k += 4) {
    v2f a, b;
    a.x = arow[k + hiK];
    a.y = arow[k + hiK + 1];
    b.x = B[(long)(k + hiK) * N + ncol];
    b.y = B[(long)(k + hiK + 1) * N + ncol];
    c = __builtin_amdgcn_wmma_f32_16x16x4_f32(false, a, false, b, (short)0, c,
                                              false, false);
  }
#else
  // 16-bit A 16x32 layout: lanes 0-15: V0..3 = K 0..7 (paired), V4..7 = K 16..23;
  // lanes 16-31: +8 / +8.  B 32x16: same K order as A (kb = kp).
  const int hi = (lane >> 4);                  // 0 or 1
  const float* arow = A + (long)mrow * K;
  for (int k = 0; k < K; k += 32) {
    v16h a, b;
    #pragma unroll
    for (int v = 0; v < 8; ++v) {
      const int kp = ((v < 4) ? (v * 2) : (16 + (v - 4) * 2)) + hi * 8;
      a[2 * v]     = (_Float16)arow[k + kp];
      a[2 * v + 1] = (_Float16)arow[k + kp + 1];
      const int kb = kp;
      b[2 * v]     = (_Float16)B[(long)(k + kb) * N + ncol];
      b[2 * v + 1] = (_Float16)B[(long)(k + kb + 1) * N + ncol];
    }
    c = __builtin_amdgcn_wmma_f32_16x16x32_f16(false, a, false, b, (short)0, c,
                                               false, false);
  }
#endif

  // C/D layout: VGPR j -> row (tileM*16 + j + 8*(lane>=16)), col = tileN*16 + lane%16.
  const int rbase = tileM * 16 + ((lane >> 4) << 3);
  #pragma unroll
  for (int j = 0; j < 8; ++j) {
    const int r = rbase + j;
    if (r < M) C[(long)r * N + ncol] = c[j];
  }
}

// deg[dst] += 1 per edge (self-loop +1 folded into dinv kernel).
__global__ void gcn_deg(const int* __restrict__ dst, float* __restrict__ deg, int E) {
  const int i = blockIdx.x * blockDim.x + threadIdx.x;
  if (i < E) atomicAdd(&deg[dst[i]], 1.0f);
}

__global__ void gcn_dinv(float* __restrict__ deg, int n) {
  const int i = blockIdx.x * blockDim.x + threadIdx.x;
  if (i < n) deg[i] = rsqrtf(deg[i] + 1.0f);
}

// One float4 feature-chunk per thread; F/4 consecutive lanes share an edge so the
// gather of h[src] is fully coalesced. Scatter via non-returning f32 atomics.
__global__ void gcn_edge_pass(const float* __restrict__ h, float* __restrict__ agg,
                              const int* __restrict__ src, const int* __restrict__ dst,
                              const float* __restrict__ dinv, int E, int F, int f4shift) {
  const int tid   = blockIdx.x * blockDim.x + threadIdx.x;
  const int total = E << f4shift;              // E * (F/4), fits in int31
  if (tid >= total) return;
  const int e = tid >> f4shift;
  const int c = tid & ((1 << f4shift) - 1);
  const int s = src[e];
  const int d = dst[e];
  const float norm = dinv[s] * dinv[d];
  const float4 v = ((const float4*)(h + (long)s * F))[c];
  float* ap = agg + (long)d * F + 4 * c;
  atomicAdd(ap + 0, v.x * norm);
  atomicAdd(ap + 1, v.y * norm);
  atomicAdd(ap + 2, v.z * norm);
  atomicAdd(ap + 3, v.w * norm);
}

// out = BN(ReLU(agg + h*dinv^2 + bias))   (elementwise; in-place safe)
__global__ void gcn_combine(const float* __restrict__ agg, const float* __restrict__ h,
                            const float* __restrict__ dinv, const float* __restrict__ b,
                            const float* __restrict__ gamma, const float* __restrict__ beta,
                            const float* __restrict__ mean, const float* __restrict__ var,
                            float* __restrict__ out, int n, int F) {
  const int i = blockIdx.x * blockDim.x + threadIdx.x;
  if (i >= n * F) return;
  const int node = i / F;
  const int f    = i - node * F;
  const float di = dinv[node];
  float t = agg[i] + h[i] * di * di + b[f];
  t = fmaxf(t, 0.0f);
  t = (t - mean[f]) * rsqrtf(var[f] + BN_EPS) * gamma[f] + beta[f];
  out[i] = t;
}

// out[n,2] = h[n,64] @ Wl[64,2] + bl
__global__ void gcn_final(const float* __restrict__ h, const float* __restrict__ Wl,
                          const float* __restrict__ bl, float* __restrict__ out, int n) {
  const int i = blockIdx.x * blockDim.x + threadIdx.x;
  if (i >= n) return;
  const float* hp = h + (long)i * 64;
  float a0 = bl[0], a1 = bl[1];
  #pragma unroll
  for (int k = 0; k < 64; ++k) {
    const float v = hp[k];
    a0 = fmaf(v, Wl[k * 2 + 0], a0);
    a1 = fmaf(v, Wl[k * 2 + 1], a1);
  }
  out[i * 2 + 0] = a0;
  out[i * 2 + 1] = a1;
}

extern "C" void kernel_launch(void* const* d_in, const int* in_sizes, int n_in,
                              void* d_out, int out_size, void* d_ws, size_t ws_size,
                              hipStream_t stream) {
  const float* x   = (const float*)d_in[0];
  const int*  eidx = (const int*)d_in[1];
  const float* W1  = (const float*)d_in[2];
  const float* b1  = (const float*)d_in[3];
  const float* W2  = (const float*)d_in[4];
  const float* b2  = (const float*)d_in[5];
  const float* g1  = (const float*)d_in[6];
  const float* be1 = (const float*)d_in[7];
  const float* m1  = (const float*)d_in[8];
  const float* v1  = (const float*)d_in[9];
  const float* g2  = (const float*)d_in[10];
  const float* be2 = (const float*)d_in[11];
  const float* m2  = (const float*)d_in[12];
  const float* v2  = (const float*)d_in[13];
  const float* Wl  = (const float*)d_in[14];
  const float* bl  = (const float*)d_in[15];

  const int n = in_sizes[0] / 256;   // 50000
  const int E = in_sizes[1] / 2;     // 1600000
  const int* src = eidx;
  const int* dst = eidx + E;

  // Workspace layout (floats): [deg/dinv : n][bufA : n*128][bufB : n*128]
  float* ws   = (float*)d_ws;
  float* dinv = ws;
  float* bufA = ws + n;                       // h1, then bn1(h1), then bn2 out
  float* bufB = bufA + (size_t)n * 128;       // agg1; later agg2 | h2
  float* agg2 = bufB;
  float* h2   = bufB + (size_t)n * 64;

  hipMemsetAsync(dinv, 0, (size_t)n * sizeof(float), stream);
  hipMemsetAsync(bufB, 0, (size_t)n * 128 * sizeof(float), stream);

  gcn_deg<<<(E + 255) / 256, 256, 0, stream>>>(dst, dinv, E);
  gcn_dinv<<<(n + 255) / 256, 256, 0, stream>>>(dinv, n);

  // Layer 1: h1 = x @ W1  (M=n, K=256, N=128) -> 8 waves/block
  gcn_gemm_wmma<<<(n + 15) / 16, 256, 0, stream>>>(x, W1, bufA, n, 256, 128);
  gcn_edge_pass<<<(E * 32 + 255) / 256, 256, 0, stream>>>(bufA, bufB, src, dst, dinv,
                                                          E, 128, 5);
  gcn_combine<<<(n * 128 + 255) / 256, 256, 0, stream>>>(bufB, bufA, dinv, b1, g1, be1,
                                                         m1, v1, bufA, n, 128);

  // Layer 2: h2 = bn1 @ W2  (M=n, K=128, N=64) -> 4 waves/block
  hipMemsetAsync(agg2, 0, (size_t)n * 64 * sizeof(float), stream);
  gcn_gemm_wmma<<<(n + 15) / 16, 128, 0, stream>>>(bufA, W2, h2, n, 128, 64);
  gcn_edge_pass<<<(E * 16 + 255) / 256, 256, 0, stream>>>(h2, agg2, src, dst, dinv,
                                                          E, 64, 4);
  gcn_combine<<<(n * 64 + 255) / 256, 256, 0, stream>>>(agg2, h2, dinv, b2, g2, be2,
                                                        m2, v2, bufA, n, 64);

  // Final linear: [n,64] @ [64,2] + bl
  gcn_final<<<(n + 255) / 256, 256, 0, stream>>>(bufA, Wl, bl, (float*)d_out, n);
}
